// BipartiteDataEncoder_35064113004568
// MI455X (gfx1250) — matrix operand order repaired and checked by reading, not compiled, not run on hardware
//
#include <hip/hip_runtime.h>
#include <hip/hip_bf16.h>

typedef __attribute__((ext_vector_type(16))) _Float16 v16h;
typedef __attribute__((ext_vector_type(8)))  float    v8f;

#define EMB 64

__device__ __forceinline__ void atomic_add_f32(float* p, float v) {
  // agent-scope relaxed FP add -> hardware global_atomic_add_f32
  __hip_atomic_fetch_add(p, v, __ATOMIC_RELAXED, __HIP_MEMORY_SCOPE_AGENT);
}

// A-fragment (16x32 f16, M x K) from an LDS tile stored row-major with given stride.
// lanes 0-15: row=lane, elems 0..15 = K0..15;
// lanes 16-31: row=lane-16, elems 0..15 = K16..31 (matches packed B order).
__device__ __forceinline__ v16h load_A_frag(const _Float16* tile, int stride, int lane, int kofs) {
  const int r  = lane & 15;
  const int kb = (lane >> 4) << 4;           // 0 or 16
  const _Float16* p = tile + r * stride + kofs + kb;
  v16h a;
#pragma unroll
  for (int j = 0; j < 8; ++j) a[j] = p[j];
#pragma unroll
  for (int j = 0; j < 8; ++j) a[8 + j] = p[8 + j];
  return a;
}

// B-fragment from pre-packed weights: [ntile][kstep][lane][16 halfs] -> 32B/lane coalesced.
__device__ __forceinline__ v16h load_B_frag(const _Float16* frag, int lane, int ntile, int kstep) {
  return *(const v16h*)(frag + (size_t)((((ntile << 1) | kstep) << 5) + lane) * 16);
}
// Single-kstep variant (K=32 weights): [ntile][lane][16 halfs].
__device__ __forceinline__ v16h load_B1_frag(const _Float16* frag, int lane, int ntile) {
  return *(const v16h*)(frag + (size_t)((ntile << 5) + lane) * 16);
}

__device__ __forceinline__ v8f wmma_f16(v16h a, v16h b, v8f c) {
  return __builtin_amdgcn_wmma_f32_16x16x32_f16(false, a, false, b, (short)0, c, false, false);
}

// Repack a 64x64 row-major f32 weight matrix into WMMA B-fragment order (f16).
// B 32x16 layout used here (contiguous k, same as load_A_frag, not the ISA table order): lanes 0-15 -> col=lane, K=0..15; lanes 16-31 -> col=lane-16, K=16..31.
__global__ void pack_wfrag_kernel(const float* __restrict__ W, _Float16* __restrict__ dst) {
  const int t    = blockIdx.x * blockDim.x + threadIdx.x;   // 0..4095
  const int j    = t & 15;
  const int lane = (t >> 4) & 31;
  const int ks   = (t >> 9) & 1;
  const int nt   = (t >> 10) & 3;
  const int K = ks * 32 + ((lane >> 4) << 4) + j;
  const int N = nt * 16 + (lane & 15);
  dst[t] = (_Float16)W[K * EMB + N];
}

// Repack a KINx64 f32 layer-1 weight into a zero-padded K=32 B-fragment (single kstep).
__global__ void pack_w1frag_kernel(const float* __restrict__ W, _Float16* __restrict__ dst, int kin) {
  const int t    = blockIdx.x * blockDim.x + threadIdx.x;   // 0..2047
  const int j    = t & 15;
  const int lane = (t >> 4) & 31;
  const int nt   = (t >> 9) & 3;
  const int K = ((lane >> 4) << 4) + j;
  const int N = nt * 16 + (lane & 15);
  dst[t] = (K < kin) ? (_Float16)W[K * EMB + N] : (_Float16)0.0f;
}

// Embedding MLP, both layers on WMMA:
//   h1 = relu((x+shift)*scale @ W1 + b1)   (K padded 19/5 -> 32, one WMMA k-step)
//   h2 = relu(h1 @ W2 + b2) [+ brk ⊗ brkW] (K=64, two WMMA k-steps)
template <int KIN, bool HASBRK>
__global__ void embed_kernel(const float* __restrict__ x,
                             const float* __restrict__ shift, const float* __restrict__ scale,
                             const _Float16* __restrict__ W1frag, const float* __restrict__ b1,
                             const _Float16* __restrict__ W2frag, const float* __restrict__ b2,
                             const float* __restrict__ brk, const float* __restrict__ brkW,
                             float* __restrict__ out, int n) {
  __shared__ _Float16 h0[256 * 32];    // 16 KB zero-padded prenormed inputs
  __shared__ _Float16 h1[256 * EMB];   // 32 KB layer-1 activations
  const int tid = threadIdx.x;
  const int row = blockIdx.x * 256 + tid;
  const int r   = row < n ? row : (n - 1);

  _Float16 tmp[32];
#pragma unroll
  for (int k = 0; k < 32; ++k)
    tmp[k] = (k < KIN) ? (_Float16)((x[(size_t)r * KIN + k] + shift[k]) * scale[k]) : (_Float16)0.0f;
  ((v16h*)&h0[tid * 32])[0] = *(const v16h*)&tmp[0];
  ((v16h*)&h0[tid * 32])[1] = *(const v16h*)&tmp[16];
  __syncthreads();

  const int wave = tid >> 5;
  const int lane = tid & 31;

  // ---- layer 1 (WMMA, K=32) ----
#pragma unroll
  for (int t = 0; t < 2; ++t) {
    const int lbase = wave * 32 + t * 16;
    const v16h a = load_A_frag(&h0[lbase * 32], 32, lane, 0);
#pragma unroll
    for (int nt = 0; nt < 4; ++nt) {
      const int col = nt * 16 + (lane & 15);
      const float bias = b1[col];
      v8f c;
#pragma unroll
      for (int i = 0; i < 8; ++i) c[i] = bias;
      c = wmma_f16(a, load_B1_frag(W1frag, lane, nt), c);
      const int rofs = lbase + ((lane >> 4) << 3);
#pragma unroll
      for (int i = 0; i < 8; ++i)
        h1[(rofs + i) * EMB + col] = (_Float16)fmaxf(c[i], 0.0f);
    }
  }
  __syncthreads();

  // ---- layer 2 (WMMA, K=64) ----
#pragma unroll
  for (int t = 0; t < 2; ++t) {
    const int lbase = wave * 32 + t * 16;
    const int gbase = blockIdx.x * 256 + lbase;
    const v16h a0 = load_A_frag(&h1[lbase * EMB], EMB, lane, 0);
    const v16h a1 = load_A_frag(&h1[lbase * EMB], EMB, lane, 32);
    const bool full = (gbase + 16 <= n);   // wave-uniform
#pragma unroll
    for (int nt = 0; nt < 4; ++nt) {
      const int col  = nt * 16 + (lane & 15);
      const float bias = b2[col];
      v8f c;
#pragma unroll
      for (int i = 0; i < 8; ++i) c[i] = bias;
      c = wmma_f16(a0, load_B_frag(W2frag, lane, nt, 0), c);
      c = wmma_f16(a1, load_B_frag(W2frag, lane, nt, 1), c);
      const int rbase = gbase + ((lane >> 4) << 3);
      if (full) {
#pragma unroll
        for (int i = 0; i < 8; ++i) {
          float v = fmaxf(c[i], 0.0f);
          if (HASBRK) v += brk[rbase + i] * brkW[col];
          out[(size_t)(rbase + i) * EMB + col] = v;
        }
      } else {
#pragma unroll
        for (int i = 0; i < 8; ++i) {
          const int rr = rbase + i;
          if (rr < n) {
            float v = fmaxf(c[i], 0.0f);
            if (HASBRK) v += brk[rr] * brkW[col];
            out[(size_t)rr * EMB + col] = v;
          }
        }
      }
    }
  }
}

// Edge aggregation: two edges per wave, float4 (b128) per half-wave lane, FP32 global atomics.
__global__ void aggregate_kernel(const float* __restrict__ xsrc,
                                 const int* __restrict__ src, const int* __restrict__ dst,
                                 float* __restrict__ sum, float* __restrict__ cnt, int nedge) {
  const unsigned gt = blockIdx.x * (unsigned)blockDim.x + threadIdx.x;
  const int wv   = (int)(gt >> 5);
  const int lane = threadIdx.x & 31;
  const int e    = wv * 2 + (lane >> 4);
  if (e >= nedge) return;
  const int fo = (lane & 15) * 4;
  const int s = src[e];
  const int d = dst[e];
  const float4 v = *(const float4*)(xsrc + (size_t)s * EMB + fo);
  float* p = sum + (size_t)d * EMB + fo;
  atomic_add_f32(p + 0, v.x);
  atomic_add_f32(p + 1, v.y);
  atomic_add_f32(p + 2, v.z);
  atomic_add_f32(p + 3, v.w);
  if ((lane & 15) == 0) atomic_add_f32(cnt + d, 1.0f);
}

// SAGE update: out = relu(mean @ Wl + bl + x_dst @ Wr). One wave per 16-row tile.
__global__ void sage_kernel(const float* __restrict__ sum, const float* __restrict__ cnt,
                            const float* __restrict__ xdst,
                            const _Float16* __restrict__ Wl, const float* __restrict__ bl,
                            const _Float16* __restrict__ Wr,
                            float* __restrict__ out, int n) {
  __shared__ _Float16 meanT[8][16 * EMB];   // 16 KB
  __shared__ _Float16 selfT[8][16 * EMB];   // 16 KB
  const int wave = threadIdx.x >> 5;
  const int lane = threadIdx.x & 31;
  const int base = (blockIdx.x * 8 + wave) * 16;

#pragma unroll 4
  for (int rr = 0; rr < 16; ++rr) {
    int row = base + rr; if (row >= n) row = n - 1;
    const float inv = 1.0f / fmaxf(cnt[row], 1.0f);
    const float2 s  = *(const float2*)(sum  + (size_t)row * EMB + lane * 2);
    const float2 xv = *(const float2*)(xdst + (size_t)row * EMB + lane * 2);
    meanT[wave][rr * EMB + lane * 2 + 0] = (_Float16)(s.x * inv);
    meanT[wave][rr * EMB + lane * 2 + 1] = (_Float16)(s.y * inv);
    selfT[wave][rr * EMB + lane * 2 + 0] = (_Float16)xv.x;
    selfT[wave][rr * EMB + lane * 2 + 1] = (_Float16)xv.y;
  }
  __syncthreads();

  const v16h am0 = load_A_frag(meanT[wave], EMB, lane, 0);
  const v16h am1 = load_A_frag(meanT[wave], EMB, lane, 32);
  const v16h ax0 = load_A_frag(selfT[wave], EMB, lane, 0);
  const v16h ax1 = load_A_frag(selfT[wave], EMB, lane, 32);
  const bool full = (base + 16 <= n);   // wave-uniform
#pragma unroll
  for (int nt = 0; nt < 4; ++nt) {
    const int col = nt * 16 + (lane & 15);
    const float bias = bl[col];
    v8f c;
#pragma unroll
    for (int i = 0; i < 8; ++i) c[i] = bias;
    c = wmma_f16(am0, load_B_frag(Wl, lane, nt, 0), c);
    c = wmma_f16(am1, load_B_frag(Wl, lane, nt, 1), c);
    c = wmma_f16(ax0, load_B_frag(Wr, lane, nt, 0), c);
    c = wmma_f16(ax1, load_B_frag(Wr, lane, nt, 1), c);
    const int rbase = base + ((lane >> 4) << 3);
    if (full) {
#pragma unroll
      for (int i = 0; i < 8; ++i)
        out[(size_t)(rbase + i) * EMB + col] = fmaxf(c[i], 0.0f);
    } else {
#pragma unroll
      for (int i = 0; i < 8; ++i) {
        const int rr = rbase + i;
        if (rr < n) out[(size_t)rr * EMB + col] = fmaxf(c[i], 0.0f);
      }
    }
  }
}

extern "C" void kernel_launch(void* const* d_in, const int* in_sizes, int n_in,
                              void* d_out, int out_size, void* d_ws, size_t ws_size,
                              hipStream_t stream) {
  const float* cons_x = (const float*)d_in[0];
  const float* var_x  = (const float*)d_in[1];
  const int*   eidx   = (const int*)d_in[2];
  const float* brk    = (const float*)d_in[4];
  const float* c_sh   = (const float*)d_in[5];
  const float* c_sc   = (const float*)d_in[6];
  const float* v_sh   = (const float*)d_in[7];
  const float* v_sc   = (const float*)d_in[8];
  const float* cW1 = (const float*)d_in[11];
  const float* cb1 = (const float*)d_in[12];
  const float* cW2 = (const float*)d_in[13];
  const float* cb2 = (const float*)d_in[14];
  const float* vW1 = (const float*)d_in[15];
  const float* vb1 = (const float*)d_in[16];
  const float* vW2 = (const float*)d_in[17];
  const float* vb2 = (const float*)d_in[18];
  const float* brkW  = (const float*)d_in[19];
  const float* Wl_cv = (const float*)d_in[20];
  const float* bl_cv = (const float*)d_in[21];
  const float* Wr_cv = (const float*)d_in[22];
  const float* Wl_vc = (const float*)d_in[23];
  const float* bl_vc = (const float*)d_in[24];
  const float* Wr_vc = (const float*)d_in[25];

  const int n_cons = in_sizes[0] / 5;
  const int n_var  = in_sizes[1] / 19;
  const int n_edge = in_sizes[2] / 2;
  const int* erow = eidx;            // cons index per edge
  const int* ecol = eidx + n_edge;   // var index per edge

  char* ws = (char*)d_ws;
  size_t off = 0;
  auto take = [&](size_t bytes) -> void* {
    void* p = ws + off;
    off = (off + bytes + 255) & ~(size_t)255;
    return p;
  };
  float* consA = (float*)take((size_t)n_cons * EMB * 4);
  float* consB = (float*)take((size_t)n_cons * EMB * 4);
  float* varA  = (float*)take((size_t)n_var  * EMB * 4);
  float* varB  = (float*)take((size_t)n_var  * EMB * 4);
  float* aggS  = (float*)take((size_t)n_var  * EMB * 4);
  float* aggC  = (float*)take((size_t)n_var  * 4);
  _Float16* wfrag  = (_Float16*)take((size_t)10 * 4096 * 2);
  _Float16* w1frag = (_Float16*)take((size_t)2 * 2048 * 2);
  (void)ws_size; (void)n_in; (void)out_size;

  _Float16* f_cW2 = wfrag;
  _Float16* f_vW2 = wfrag + 4096;
  _Float16* f_Wl_cv[2] = { wfrag + 2 * 4096, wfrag + 3 * 4096 };
  _Float16* f_Wr_cv[2] = { wfrag + 4 * 4096, wfrag + 5 * 4096 };
  _Float16* f_Wl_vc[2] = { wfrag + 6 * 4096, wfrag + 7 * 4096 };
  _Float16* f_Wr_vc[2] = { wfrag + 8 * 4096, wfrag + 9 * 4096 };
  _Float16* f_cW1 = w1frag;
  _Float16* f_vW1 = w1frag + 2048;

  dim3 b256(256);
  pack_wfrag_kernel<<<16, b256, 0, stream>>>(cW2, f_cW2);
  pack_wfrag_kernel<<<16, b256, 0, stream>>>(vW2, f_vW2);
  for (int l = 0; l < 2; ++l) {
    pack_wfrag_kernel<<<16, b256, 0, stream>>>(Wl_cv + l * 4096, f_Wl_cv[l]);
    pack_wfrag_kernel<<<16, b256, 0, stream>>>(Wr_cv + l * 4096, f_Wr_cv[l]);
    pack_wfrag_kernel<<<16, b256, 0, stream>>>(Wl_vc + l * 4096, f_Wl_vc[l]);
    pack_wfrag_kernel<<<16, b256, 0, stream>>>(Wr_vc + l * 4096, f_Wr_vc[l]);
  }
  pack_w1frag_kernel<<<8, b256, 0, stream>>>(cW1, f_cW1, 5);
  pack_w1frag_kernel<<<8, b256, 0, stream>>>(vW1, f_vW1, 19);

  embed_kernel<5, false><<<(n_cons + 255) / 256, b256, 0, stream>>>(
      cons_x, c_sh, c_sc, f_cW1, cb1, f_cW2, cb2, nullptr, nullptr, consA, n_cons);
  embed_kernel<19, true><<<(n_var + 255) / 256, b256, 0, stream>>>(
      var_x, v_sh, v_sc, f_vW1, vb1, f_vW2, vb2, brk, brkW, varA, n_var);

  const float* ccur = consA;
  const float* vcur = varA;
  const int aggBlocks = (n_edge + 15) / 16;   // 8 waves x 2 edges per 256-thread block

  for (int l = 0; l < 2; ++l) {
    // cons -> var
    hipMemsetAsync(aggS, 0, (size_t)n_var * EMB * 4, stream);
    hipMemsetAsync(aggC, 0, (size_t)n_var * 4, stream);
    aggregate_kernel<<<aggBlocks, b256, 0, stream>>>(ccur, erow, ecol, aggS, aggC, n_edge);
    float* vnew = (l == 1) ? (float*)d_out : varB;
    sage_kernel<<<(n_var + 127) / 128, b256, 0, stream>>>(
        aggS, aggC, vcur, f_Wl_cv[l], bl_cv + l * EMB, f_Wr_cv[l], vnew, n_var);

    // var -> cons (reads the OLD var, matching HeteroConv semantics)
    hipMemsetAsync(aggS, 0, (size_t)n_cons * EMB * 4, stream);
    hipMemsetAsync(aggC, 0, (size_t)n_cons * 4, stream);
    aggregate_kernel<<<aggBlocks, b256, 0, stream>>>(vcur, ecol, erow, aggS, aggC, n_edge);
    float* cnew = (l == 0) ? consB : consA;
    sage_kernel<<<(n_cons + 127) / 128, b256, 0, stream>>>(
        aggS, aggC, ccur, f_Wl_vc[l], bl_vc + l * EMB, f_Wr_vc[l], cnew, n_cons);

    ccur = cnew;
    vcur = vnew;
  }
}
